// Model_7361573945762
// MI455X (gfx1250) — matrix operand order repaired and checked by reading, not compiled, not run on hardware
//
#include <hip/hip_runtime.h>
#include <hip/hip_bf16.h>

// ---------- gfx1250 async-to-LDS availability (compile-safe guards) ----------
#if defined(__has_builtin)
#  if __has_builtin(__builtin_amdgcn_global_load_async_to_lds_b128)
#    define HAS_ASYNC_LDS 1
#  endif
#endif
#ifndef HAS_ASYNC_LDS
#  define HAS_ASYNC_LDS 0
#endif

typedef int v4i_t __attribute__((vector_size(16)));

__device__ __forceinline__ void wait_async0() {
#if HAS_ASYNC_LDS
#  if __has_builtin(__builtin_amdgcn_s_wait_asynccnt)
    __builtin_amdgcn_s_wait_asynccnt(0);
#  else
    asm volatile("s_wait_asynccnt 0x0" ::: "memory");
#  endif
#endif
}

__device__ __forceinline__ void async_cp16(const void* gsrc, void* ldst) {
#if HAS_ASYNC_LDS
    __builtin_amdgcn_global_load_async_to_lds_b128(
        (__attribute__((address_space(1))) v4i_t*)(void*)gsrc,
        (__attribute__((address_space(3))) v4i_t*)ldst, 0, 0);
#else
    *(uint4*)ldst = *(const uint4*)gsrc;
#endif
}

// ---------- WMMA types (CDNA5 / gfx1250, wave32) ----------
typedef __attribute__((ext_vector_type(16))) __bf16 v16bf;
typedef __attribute__((ext_vector_type(8)))  float  v8f;

union FragBF { uint4 q[2]; v16bf v; };   // 32 bytes

__device__ __forceinline__ unsigned f2bf_rne(float f) {
    unsigned x = __float_as_uint(f);
    unsigned r = x + 0x7FFFu + ((x >> 16) & 1u);   // round-to-nearest-even
    return r >> 16;
}
__device__ __forceinline__ unsigned pack_bf2(float lo, float hi) {
    return f2bf_rne(lo) | (f2bf_rne(hi) << 16);
}

// ---------- weight prep: f32 [cnt][K][N] -> bf16 transposed [cnt][N][K] ----------
__global__ void wt_transpose_bf16(const float* __restrict__ W, unsigned short* __restrict__ Wt,
                                  int K, int N, long total) {
    long stride = (long)gridDim.x * blockDim.x;
    for (long o = (long)blockIdx.x * blockDim.x + threadIdx.x; o < total; o += stride) {
        long mat = o / ((long)K * N);
        long r   = o - mat * (long)K * N;
        int  n   = (int)(r / K);
        int  k   = (int)(r - (long)n * K);
        Wt[o] = (unsigned short)f2bf_rne(W[mat * (long)K * N + (size_t)k * N + n]);
    }
}

// ---------- segment-mean helpers ----------
__global__ void deg_count(const int* __restrict__ dst, float* __restrict__ deg, int E) {
    for (int i = blockIdx.x * blockDim.x + threadIdx.x; i < E; i += gridDim.x * blockDim.x)
        atomicAdd(&deg[dst[i]], 1.0f);
}

__global__ void inv_deg(float* deg, int n) {
    int i = blockIdx.x * blockDim.x + threadIdx.x;
    if (i < n) deg[i] = 1.0f / fmaxf(deg[i], 1.0f);
}

// mean directly: atomicAdd(x_src[src] * invdeg[dst]); consecutive threads cover one row
__global__ void scatter_mean4(const float* __restrict__ xsrc, const int* __restrict__ src,
                              const int* __restrict__ dst, const float* __restrict__ invdeg,
                              float* __restrict__ msum, long total, int f4shift, int f4mask) {
    for (long i = (long)blockIdx.x * blockDim.x + threadIdx.x; i < total;
         i += (long)gridDim.x * blockDim.x) {
        int e = (int)(i >> f4shift);
        int c = (int)(i & f4mask);
        int s = src[e], d = dst[e];
        float sc = invdeg[d];
        const float4 v = ((const float4*)xsrc)[((size_t)s << f4shift) + c];
        float* o = msum + ((((size_t)d << f4shift) + c) << 2);
        atomicAdd(o + 0, v.x * sc); atomicAdd(o + 1, v.y * sc);
        atomicAdd(o + 2, v.z * sc); atomicAdd(o + 3, v.w * sc);
    }
}

// ---------- fused multi-pair WMMA GEMM ----------
// C[M,N] = relu?( sum_p A_p @ B_p^T-stored + bias1 + bias2 )
// A_p: f32 [M,K] (row stride == K), optional per-row gather idx
// Bt_p: bf16 [N][K] (weights pre-transposed+converted once per launch)
// Requires K%32==0, N%128==0 (K in {128,256}, N=256 here).
struct GemmPair { const float* A; const unsigned short* Bt; const int* idx; };
struct GemmArgs {
    GemmPair p[4];
    const float* bias1; const float* bias2;
    float* C;
    int M, K, N, npairs, relu;
};

#define TM 128
#define TN 128
#define TK 32
#define LDW 40           // (TK + 8) bf16 per row: 80B stride -> conflict-free b128 reads

__launch_bounds__(256)
__global__ void sage_gemm(GemmArgs g) {
    // bf16 tiles in natural K order (A [row][k]); B held transposed [col][k]
    __shared__ __align__(16) unsigned short As16[TM * LDW];
    __shared__ __align__(16) unsigned short Bs16[TN * LDW];

    const int tid  = threadIdx.x;
    const int lane = tid & 31;
    const int wave = tid >> 5;          // 8 waves
    const int wm   = wave >> 1;         // 0..3 (M dir, 32 rows each)
    const int wn   = wave & 1;          // 0..1 (N dir, 64 cols each)
    const int half = lane >> 4;         // 0/1
    const int l15  = lane & 15;

    const int mBase = blockIdx.y * TM;
    const int nBase = blockIdx.x * TN;

    v8f acc[2][4];
    #pragma unroll
    for (int t = 0; t < 2; ++t)
        #pragma unroll
        for (int u = 0; u < 4; ++u)
            acc[t][u] = v8f{};

    unsigned* As32 = (unsigned*)As16;
    const uint4* As4 = (const uint4*)As16;
    const uint4* Bs4 = (const uint4*)Bs16;

    for (int p = 0; p < g.npairs; ++p) {
        const float*          A   = g.p[p].A;
        const unsigned short* Bt  = g.p[p].Bt;
        const int*            idx = g.p[p].idx;

        for (int kb = 0; kb < g.K; kb += TK) {
            __syncthreads();
            // ---- stage B tile: raw 16B/lane async copies of pre-converted bf16 [N][K] ----
            #pragma unroll
            for (int i = 0; i < 2; ++i) {
                int q   = tid + i * 256;        // 0..511 : 128 cols x 4 x 16B segs
                int nl  = q >> 2;
                int seg = q & 3;
                const unsigned short* src = Bt + (size_t)(nBase + nl) * g.K + kb + seg * 8;
                unsigned short*       dst = Bs16 + nl * LDW + seg * 8;
                async_cp16(src, dst);
            }
            // ---- stage A tile: TM x TK f32 -> packed bf16, 4 x float4 per thread ----
            #pragma unroll
            for (int i = 0; i < 4; ++i) {
                int f4  = tid + i * 256;          // 0..1023
                int row = f4 >> 3;                // 8 float4 per 32-wide row
                int c4  = (f4 & 7) << 2;
                int grow = mBase + row;
                float4 v = make_float4(0.f, 0.f, 0.f, 0.f);
                if (grow < g.M) {
                    int arow = idx ? idx[grow] : grow;
                    v = *(const float4*)(A + (size_t)arow * g.K + kb + c4);
                    if (kb + TK < g.K)
                        __builtin_prefetch(A + (size_t)arow * g.K + kb + TK + c4, 0, 1);
                }
                As32[row * (LDW / 2) + (c4 >> 1)]     = pack_bf2(v.x, v.y);
                As32[row * (LDW / 2) + (c4 >> 1) + 1] = pack_bf2(v.z, v.w);
            }
            wait_async0();
            __syncthreads();

            // ---- fragments: pure b128 LDS loads (layouts already match ISA) ----
            FragBF afr[2], bfr[4];
            #pragma unroll
            for (int t = 0; t < 2; ++t) {
                int r = wm * 32 + t * 16 + l15;
                int b = r * (LDW / 8) + half;     // 16B units: r*5 + half
                afr[t].q[0] = As4[b];             // K k0..k0+7   (2 per VGPR)
                afr[t].q[1] = As4[b + 2];         // K k0+16..k0+23
            }
            #pragma unroll
            for (int u = 0; u < 4; ++u) {
                int c = wn * 64 + u * 16 + l15;
                int b = c * (LDW / 8) + half;     // 16B units: c*5 + half
                bfr[u].q[0] = Bs4[b];             // K half*8 .. +7
                bfr[u].q[1] = Bs4[b + 2];         // K 16+half*8 .. +7
            }
            #pragma unroll
            for (int t = 0; t < 2; ++t)
                #pragma unroll
                for (int u = 0; u < 4; ++u)
                    acc[t][u] = __builtin_amdgcn_wmma_f32_16x16x32_bf16(
                        false, afr[t].v, false, bfr[u].v,
                        (short)0, acc[t][u], false, false);
        }
    }

    // ---- epilogue: bias + relu + store (C/D layout: VGPR r -> M=r / r+8) ----
    #pragma unroll
    for (int u = 0; u < 4; ++u) {
        int col = nBase + wn * 64 + u * 16 + l15;
        float b = 0.f;
        if (g.bias1) b += g.bias1[col];
        if (g.bias2) b += g.bias2[col];
        #pragma unroll
        for (int t = 0; t < 2; ++t) {
            #pragma unroll
            for (int r = 0; r < 8; ++r) {
                int row = mBase + wm * 32 + t * 16 + half * 8 + r;
                if (row < g.M) {
                    float v = acc[t][u][r] + b;
                    if (g.relu) v = fmaxf(v, 0.f);
                    g.C[(size_t)row * g.N + col] = v;
                }
            }
        }
    }
}

// ---------- decoder final dot: out[e] = h[e,:] . W2 + b2 ----------
__global__ void dec_out(const float* __restrict__ h, const float* __restrict__ W2,
                        const float* __restrict__ b2, float* __restrict__ out,
                        int EL, int C) {
    int e    = blockIdx.x * (blockDim.x >> 5) + (threadIdx.x >> 5);
    int lane = threadIdx.x & 31;
    if (e >= EL) return;
    float s = 0.f;
    for (int c = lane; c < C; c += 32) s += h[(size_t)e * C + c] * W2[c];
    #pragma unroll
    for (int o = 16; o; o >>= 1) s += __shfl_down(s, o, 32);
    if (lane == 0) out[e] = s + b2[0];
}

// ---------------------------------------------------------------------------
extern "C" void kernel_launch(void* const* d_in, const int* in_sizes, int n_in,
                              void* d_out, int out_size, void* d_ws, size_t ws_size,
                              hipStream_t stream) {
    const float* x_author = (const float*)d_in[0];
    const float* x_paper  = (const float*)d_in[1];
    const float* Wl1 = (const float*)d_in[2];
    const float* bl1 = (const float*)d_in[3];
    const float* Wr1 = (const float*)d_in[4];
    const float* Wl  = (const float*)d_in[5];
    const float* bl  = (const float*)d_in[6];
    const float* Wr  = (const float*)d_in[7];
    const float* dW1 = (const float*)d_in[8];
    const float* db1 = (const float*)d_in[9];
    const float* dW2 = (const float*)d_in[10];
    const float* db2 = (const float*)d_in[11];
    const int* wsrc = (const int*)d_in[12];
    const int* wdst = (const int*)d_in[13];
    const int* csrc = (const int*)d_in[14];
    const int* cdst = (const int*)d_in[15];
    const int* esrc = (const int*)d_in[16];
    const int* edst = (const int*)d_in[17];

    const int C  = in_sizes[3] / 3;            // 256
    const int F  = in_sizes[2] / (3 * C);      // 128
    const int Na = in_sizes[0] / F;            // 50000
    const int Np = in_sizes[1] / F;            // 100000
    const int E  = in_sizes[12];               // 800000
    const int EL = in_sizes[16];               // 100000

    // ---- workspace carve-up ----
    float* w = (float*)d_ws;
    size_t off = 0;
    auto alloc = [&](size_t n) { float* p = w + off; off += n; return p; };
    float* xa0   = alloc((size_t)Na * C);
    float* xa1   = alloc((size_t)Na * C);
    float* xp0   = alloc((size_t)Np * C);
    float* xp1   = alloc((size_t)Np * C);
    float* mean0 = alloc((size_t)Np * C);      // also reused as decoder h
    float* mean1 = alloc((size_t)Np * C);
    float* deg_wp = alloc(Np);
    float* deg_cp = alloc(Np);
    float* deg_wa = alloc(Na);
    // bf16 transposed weight pool
    const size_t nWl1 = (size_t)3 * F * C, nWl = (size_t)9 * C * C, nD = (size_t)2 * C * C;
    unsigned short* pool = (unsigned short*)(w + off);
    unsigned short* Wl1t = pool;
    unsigned short* Wr1t = Wl1t + nWl1;
    unsigned short* Wlt  = Wr1t + nWl1;
    unsigned short* Wrt  = Wlt + nWl;
    unsigned short* dW1t = Wrt + nWl;
    off += (2 * nWl1 + 2 * nWl + nD + 1) / 2;
    (void)ws_size;

    // ---- one-time weight transpose+convert (f32 [K][N] -> bf16 [N][K]) ----
    wt_transpose_bf16<<<512, 256, 0, stream>>>(Wl1, Wl1t, F, C, (long)nWl1);
    wt_transpose_bf16<<<512, 256, 0, stream>>>(Wr1, Wr1t, F, C, (long)nWl1);
    wt_transpose_bf16<<<512, 256, 0, stream>>>(Wl,  Wlt,  C, C, (long)nWl);
    wt_transpose_bf16<<<512, 256, 0, stream>>>(Wr,  Wrt,  C, C, (long)nWl);
    wt_transpose_bf16<<<512, 256, 0, stream>>>(dW1, dW1t, C, C, (long)nD);

    // ---- inverse degrees (layer-invariant) ----
    (void)hipMemsetAsync(deg_wp, 0, (size_t)Np * 4, stream);
    (void)hipMemsetAsync(deg_cp, 0, (size_t)Np * 4, stream);
    (void)hipMemsetAsync(deg_wa, 0, (size_t)Na * 4, stream);
    deg_count<<<1024, 256, 0, stream>>>(wdst, deg_wp, E);
    deg_count<<<1024, 256, 0, stream>>>(cdst, deg_cp, E);
    deg_count<<<1024, 256, 0, stream>>>(wsrc, deg_wa, E);
    inv_deg<<<(Np + 255) / 256, 256, 0, stream>>>(deg_wp, Np);
    inv_deg<<<(Np + 255) / 256, 256, 0, stream>>>(deg_cp, Np);
    inv_deg<<<(Na + 255) / 256, 256, 0, stream>>>(deg_wa, Na);

    auto run_layer = [&](const float* cura, const float* curp, float* nexta, float* nextp,
                         int Kf, const unsigned short* WlLt, const float* blL,
                         const unsigned short* WrLt, int relu) {
        const int f4   = Kf >> 2;
        const int f4sh = (f4 == 32) ? 5 : 6;
        const int f4mk = f4 - 1;
        const size_t wfc = (size_t)Kf * C;     // per-edge-type weight block (elements)

        // -- author side: mean over reverse-writes of paper feats --
        (void)hipMemsetAsync(mean0, 0, (size_t)Na * Kf * 4, stream);
        scatter_mean4<<<4096, 256, 0, stream>>>(curp, wdst, wsrc, deg_wa, mean0,
                                                (long)E * f4, f4sh, f4mk);
        {
            GemmArgs g{};
            g.p[0] = { mean0, WlLt + 1 * wfc, nullptr };
            g.p[1] = { cura,  WrLt + 1 * wfc, nullptr };
            g.bias1 = blL + 1 * C; g.bias2 = nullptr;
            g.C = nexta; g.M = Na; g.K = Kf; g.N = C; g.npairs = 2; g.relu = relu;
            sage_gemm<<<dim3(C / TN, (Na + TM - 1) / TM), 256, 0, stream>>>(g);
        }
        // -- paper side: writes-mean + cites-mean + 2 root terms --
        (void)hipMemsetAsync(mean0, 0, (size_t)Np * Kf * 4, stream);
        (void)hipMemsetAsync(mean1, 0, (size_t)Np * Kf * 4, stream);
        scatter_mean4<<<4096, 256, 0, stream>>>(cura, wsrc, wdst, deg_wp, mean0,
                                                (long)E * f4, f4sh, f4mk);
        scatter_mean4<<<4096, 256, 0, stream>>>(curp, csrc, cdst, deg_cp, mean1,
                                                (long)E * f4, f4sh, f4mk);
        {
            GemmArgs g{};
            g.p[0] = { mean0, WlLt + 0 * wfc, nullptr };
            g.p[1] = { mean1, WlLt + 2 * wfc, nullptr };
            g.p[2] = { curp,  WrLt + 0 * wfc, nullptr };
            g.p[3] = { curp,  WrLt + 2 * wfc, nullptr };
            g.bias1 = blL + 0 * C; g.bias2 = blL + 2 * C;
            g.C = nextp; g.M = Np; g.K = Kf; g.N = C; g.npairs = 4; g.relu = relu;
            sage_gemm<<<dim3(C / TN, (Np + TM - 1) / TM), 256, 0, stream>>>(g);
        }
    };

    // layer 1 (F->C) then 3 layers (C->C); relu after all but last
    run_layer(x_author, x_paper, xa0, xp0, F, Wl1t, bl1, Wr1t, 1);
    const size_t lcc = (size_t)3 * C * C, lc = (size_t)3 * C;
    run_layer(xa0, xp0, xa1, xp1, C, Wlt + 0 * lcc, bl + 0 * lc, Wrt + 0 * lcc, 1);
    run_layer(xa1, xp1, xa0, xp0, C, Wlt + 1 * lcc, bl + 1 * lc, Wrt + 1 * lcc, 1);
    run_layer(xa0, xp0, xa1, xp1, C, Wlt + 2 * lcc, bl + 2 * lc, Wrt + 2 * lcc, 0);
    // final author embedding is in xa1

    // decoder: h = relu(xa[src]@W1_top + xa[dst]@W1_bot + b1), reuse mean0 for h
    {
        GemmArgs g{};
        g.p[0] = { xa1, dW1t,                 esrc };
        g.p[1] = { xa1, dW1t + (size_t)C * C, edst };
        g.bias1 = db1; g.bias2 = nullptr;
        g.C = mean0; g.M = EL; g.K = C; g.N = C; g.npairs = 2; g.relu = 1;
        sage_gemm<<<dim3(C / TN, (EL + TM - 1) / TM), 256, 0, stream>>>(g);
    }
    dec_out<<<(EL + 7) / 8, 256, 0, stream>>>(mean0, dW2, db2, (float*)d_out, EL, C);
}
